// Model_13915694039432
// MI455X (gfx1250) — matrix operand order repaired and checked by reading, not compiled, not run on hardware
//
#include <hip/hip_runtime.h>
#include <hip/hip_bf16.h>

typedef __bf16 bf16;
typedef __attribute__((ext_vector_type(16))) __bf16 v16bf;
typedef __attribute__((ext_vector_type(8)))  __bf16 v8bf;
typedef __attribute__((ext_vector_type(8)))  float  v8f;

#define B_   16
#define S_   512
#define H_   1024
#define V_   8000
#define L_   8
#define HEADS_ 4
#define DH_  256
#define M_   (B_ * S_)      // 8192 token rows
#define NEGINF (-1e30f)
#define LDK  40             // padded LDS K-stride (elements)

static __device__ __forceinline__ void wait_async0() {
#if __has_builtin(__builtin_amdgcn_s_wait_asynccnt)
    __builtin_amdgcn_s_wait_asynccnt(0);
#else
    asm volatile("s_wait_asynccnt 0x0" ::: "memory");
#endif
}

// ---------------------------------------------------------------------------
// f32 -> bf16 weight conversion (run once per call into the ws bf16 pool)
// ---------------------------------------------------------------------------
__global__ __launch_bounds__(256) void cvt_bf16_kernel(
    const float* __restrict__ src, bf16* __restrict__ dst, long long n)
{
    long long i = (long long)blockIdx.x * 256 + threadIdx.x;
    if (i < n) dst[i] = (bf16)src[i];
}

// ---------------------------------------------------------------------------
// Embedding + sinusoidal positional encoding: z[row][c] = emb[x[row]][c]+pe
// ---------------------------------------------------------------------------
__global__ __launch_bounds__(256) void embed_pe_kernel(
    const int* __restrict__ x, const float* __restrict__ emb, float* __restrict__ z)
{
    int row = blockIdx.x;                // 0..M_-1
    int s   = row & (S_ - 1);
    int tok = x[row];
    int tid = threadIdx.x;
    #pragma unroll
    for (int i = 0; i < 4; ++i) {
        int c = tid + 256 * i;           // 0..1023
        int pair = c >> 1;
        float freq = __expf(-(float)(2 * pair) * (9.210340371976184f / (float)H_));
        float ang  = (float)s * freq;
        float pe   = (c & 1) ? __cosf(ang) : __sinf(ang);
        z[(size_t)row * H_ + c] = emb[(size_t)tok * H_ + c] + pe;
    }
}

// ---------------------------------------------------------------------------
// bf16-WMMA GEMM with pre-converted bf16 weights:
//   Y = act(X @ W^T + bias),  X f32 [M x K] (ldx), W bf16 [N x K] dense,
//   Y f32 [M x N] (ldy).
// Block tile 128(M) x 64(N), 128 threads = 4 waves; wave owns 32 M-rows
// (2 A fragments) so each B fragment feeds 2 WMMAs.
// W tile staged via CDNA5 async LDS loads (ASYNCcnt, VGPR-bypassing);
// X tile staged row-per-thread with packed bf16 conversion + b128 DS stores.
// M % 128 == 0, N % 64 == 0, K % 32 == 0 in all calls.
// ---------------------------------------------------------------------------
__global__ __launch_bounds__(128) void gemm_bf16w_kernel(
    const float* __restrict__ X, const bf16* __restrict__ W,
    const float* __restrict__ bias, float* __restrict__ Y,
    int M, int N, int K, int ldx, int ldy, int act)
{
    __shared__ bf16 smem[(128 + 64) * LDK];
    bf16* Xs = smem;                       // [128][LDK] at LDS offset 0
    bf16* Ws = smem + 128 * LDK;           // [64][LDK]
    const unsigned WS_LDS_BASE = (unsigned)(128 * LDK * sizeof(bf16));

    int tid  = threadIdx.x;
    int wave = tid >> 5;
    int lane = tid & 31;
    int bm = blockIdx.y * 128, bn = blockIdx.x * 64;
    int mloc = lane & 15;
    int kh   = lane >> 4;

    v8f vzero = {0.f, 0.f, 0.f, 0.f, 0.f, 0.f, 0.f, 0.f};
    v8f acc0[4] = {vzero, vzero, vzero, vzero};
    v8f acc1[4] = {vzero, vzero, vzero, vzero};

    for (int k0 = 0; k0 < K; k0 += 32) {
        // ---- async-stage W tile (64 rows x 32 bf16 = 256 x 16B chunks)
        #pragma unroll
        for (int it = 0; it < 2; ++it) {
            int chunk = it * 128 + tid;          // 0..255
            int r = chunk >> 2, cq = chunk & 3;  // row, 16B quarter
            unsigned ldsa = WS_LDS_BASE + (unsigned)((r * LDK + cq * 8) * sizeof(bf16));
            const bf16* gp = W + (size_t)(bn + r) * K + k0 + cq * 8;
            asm volatile("global_load_async_to_lds_b128 %0, %1, off"
                         :: "v"(ldsa), "v"((unsigned long long)(size_t)gp)
                         : "memory");
        }
        // ---- stage X tile: one 32-element row per thread, vector DS stores
        {
            const float* src = X + (size_t)(bm + tid) * ldx + k0;
            v8bf* dst = (v8bf*)(Xs + tid * LDK);
            #pragma unroll
            for (int qd = 0; qd < 4; ++qd) {
                v8bf t;
                #pragma unroll
                for (int j = 0; j < 8; ++j) t[j] = (bf16)src[qd * 8 + j];
                dst[qd] = t;
            }
        }
        wait_async0();
        __syncthreads();

        // ---- A fragments: rows 32*wave+mloc and +16, contiguous k = 16*kh + j
        int rowA = 32 * wave + mloc;
        v16bf a0, a1;
        #pragma unroll
        for (int j = 0; j < 8; ++j) {
            a0[j]     = Xs[rowA * LDK + 16 * kh + j];
            a0[j + 8] = Xs[rowA * LDK + 16 * kh + 8 + j];
            a1[j]     = Xs[(rowA + 16) * LDK + 16 * kh + j];
            a1[j + 8] = Xs[(rowA + 16) * LDK + 16 * kh + 8 + j];
        }
        // ---- 4 B fragments, each reused by 2 WMMAs
        #pragma unroll
        for (int t = 0; t < 4; ++t) {
            v16bf bfrag;
            #pragma unroll
            for (int j = 0; j < 16; ++j)
                bfrag[j] = Ws[(16 * t + mloc) * LDK + 16 * kh + j];
            acc0[t] = __builtin_amdgcn_wmma_f32_16x16x32_bf16(
                false, a0, false, bfrag, (short)0, acc0[t], false, false);
            acc1[t] = __builtin_amdgcn_wmma_f32_16x16x32_bf16(
                false, a1, false, bfrag, (short)0, acc1[t], false, false);
        }
        __syncthreads();
    }

    // ---- epilogue: bias + optional exact GELU, C/D layout M = r + 8*kh
    #pragma unroll
    for (int half = 0; half < 2; ++half) {
        int mbase = bm + 32 * wave + 16 * half + 8 * kh;
        #pragma unroll
        for (int t = 0; t < 4; ++t) {
            int n = bn + 16 * t + mloc;
            float bv = bias ? bias[n] : 0.f;
            #pragma unroll
            for (int r = 0; r < 8; ++r) {
                float val = (half ? acc1[t][r] : acc0[t][r]) + bv;
                if (act == 1)
                    val = 0.5f * val * (1.f + erff(val * 0.70710678118654752f));
                Y[(size_t)(mbase + r) * ldy + n] = val;
            }
        }
    }
}

// ---------------------------------------------------------------------------
// Batched o = p @ v GEMM (W = f32 activations [K x N] with row stride ldw).
// Batch z decomposes as (outer=b, inner=h) with independent pointer strides.
// ---------------------------------------------------------------------------
__global__ __launch_bounds__(128) void gemm_pv_kernel(
    const float* __restrict__ X, const float* __restrict__ W,
    float* __restrict__ Y,
    int M, int N, int K, int ldx, int ldw, int ldy, int batchInner,
    long long sXo, long long sXi, long long sWo, long long sWi,
    long long sYo, long long sYi)
{
    int bz = blockIdx.z;
    int zo = bz / batchInner, zi = bz % batchInner;
    X += zo * sXo + zi * sXi;
    W += zo * sWo + zi * sWi;
    Y += zo * sYo + zi * sYi;

    __shared__ bf16 Xs[64][LDK];
    __shared__ bf16 Ws[64][LDK];

    int tid  = threadIdx.x;
    int wave = tid >> 5;
    int lane = tid & 31;
    int bm = blockIdx.y * 64, bn = blockIdx.x * 64;
    int mloc = lane & 15;
    int kh   = lane >> 4;

    v8f vzero = {0.f, 0.f, 0.f, 0.f, 0.f, 0.f, 0.f, 0.f};
    v8f acc[4] = {vzero, vzero, vzero, vzero};

    for (int k0 = 0; k0 < K; k0 += 32) {
        // X: one 32-element row per 2 threads would misalign; do row/half per thread
        {
            int r = tid >> 1, hf = tid & 1;           // 64 rows x 2 halves
            const float* src = X + (size_t)(bm + r) * ldx + k0 + hf * 16;
            v8bf* dst = (v8bf*)(&Xs[r][hf * 16]);
            #pragma unroll
            for (int qd = 0; qd < 2; ++qd) {
                v8bf t;
                #pragma unroll
                for (int j = 0; j < 8; ++j) t[j] = (bf16)src[qd * 8 + j];
                dst[qd] = t;
            }
        }
        // W staged transposed (scattered reads, unavoidable for [K x N] layout)
        for (int idx = tid; idx < 64 * 32; idx += 128) {
            int r = idx >> 5, c = idx & 31;
            Ws[r][c] = (bf16)W[(size_t)(k0 + c) * ldw + bn + r];
        }
        __syncthreads();

        v16bf a;
        #pragma unroll
        for (int j = 0; j < 8; ++j) {
            a[j]     = Xs[16 * wave + mloc][16 * kh + j];
            a[j + 8] = Xs[16 * wave + mloc][16 * kh + 8 + j];
        }
        #pragma unroll
        for (int t = 0; t < 4; ++t) {
            v16bf bfrag;
            #pragma unroll
            for (int j = 0; j < 16; ++j)
                bfrag[j] = Ws[16 * t + mloc][16 * kh + j];
            acc[t] = __builtin_amdgcn_wmma_f32_16x16x32_bf16(
                false, a, false, bfrag, (short)0, acc[t], false, false);
        }
        __syncthreads();
    }

    int mbase = bm + 16 * wave + 8 * kh;
    #pragma unroll
    for (int t = 0; t < 4; ++t) {
        int n = bn + 16 * t + mloc;
        #pragma unroll
        for (int r = 0; r < 8; ++r)
            Y[(size_t)(mbase + r) * ldy + n] = acc[t][r];
    }
}

// ---------------------------------------------------------------------------
// Attention scores + masked softmax.  One 128-thread block per (b, head, qt16).
// s = (q @ k^T) / sqrt(H); mask = causal & (q<len) & (head<len quirk);
// p -> global [B,H,S,S]; last layer also writes atten_last = max(p, axis=-1).
// ---------------------------------------------------------------------------
__global__ __launch_bounds__(128) void attn_scores_kernel(
    const float* __restrict__ q, const float* __restrict__ k,
    const int* __restrict__ lengths, float* __restrict__ p,
    float* __restrict__ attn_out, int is_last)
{
    int qt = blockIdx.x;     // 0..31  (16-query tile)
    int h  = blockIdx.y;     // 0..3
    int b  = blockIdx.z;     // 0..15

    __shared__ float sc[16][S_];   // 32 KB score tile

    int tid  = threadIdx.x;
    int wave = tid >> 5;
    int lane = tid & 31;
    int mloc = lane & 15;
    int kh   = lane >> 4;
    const float scale = 0.03125f;  // 1/sqrt(1024)

    // preload Q A-fragments for full dh=256 (8 K-steps of 32)
    const size_t qrowbase = ((size_t)(b * S_ + qt * 16 + mloc)) * H_ + (size_t)h * DH_;
    v16bf aq[8];
    #pragma unroll
    for (int kc = 0; kc < 8; ++kc) {
        #pragma unroll
        for (int j = 0; j < 8; ++j) {
            aq[kc][j]     = (bf16)q[qrowbase + 32 * kc + 16 * kh + j];
            aq[kc][j + 8] = (bf16)q[qrowbase + 32 * kc + 16 * kh + 8 + j];
        }
    }
    v8f vzero = {0.f, 0.f, 0.f, 0.f, 0.f, 0.f, 0.f, 0.f};

    // each wave covers 8 key tiles of 16
    for (int i = 0; i < 8; ++i) {
        int kt = wave * 8 + i;
        v8f acc = vzero;
        #pragma unroll
        for (int kc = 0; kc < 8; ++kc) {
            const size_t krowbase =
                ((size_t)(b * S_ + kt * 16 + mloc)) * H_ + (size_t)h * DH_;
            v16bf bk;
            #pragma unroll
            for (int j = 0; j < 16; ++j)
                bk[j] = (bf16)k[krowbase + 32 * kc + 16 * kh + j];
            acc = __builtin_amdgcn_wmma_f32_16x16x32_bf16(
                false, aq[kc], false, bk, (short)0, acc, false, false);
        }
        #pragma unroll
        for (int r = 0; r < 8; ++r)
            sc[r + 8 * kh][kt * 16 + mloc] = acc[r] * scale;
    }
    __syncthreads();

    // masked softmax: thread t (<16) owns query row t
    if (tid < 16) {
        int row  = tid;
        int qrow = qt * 16 + row;
        int len  = lengths[b];
        bool row_valid = (qrow < len) && (h < len);
        float mx = NEGINF;
        for (int c = 0; c < S_; ++c) {
            bool valid = row_valid && (c <= qrow);
            float s = valid ? sc[row][c] : NEGINF;
            sc[row][c] = s;
            mx = fmaxf(mx, s);
        }
        float sum = 0.f;
        for (int c = 0; c < S_; ++c) {
            float e = __expf(sc[row][c] - mx);
            sc[row][c] = e;
            sum += e;
        }
        float inv = 1.f / sum;
        float pmax = 0.f;
        for (int c = 0; c < S_; ++c) {
            float pv = row_valid ? sc[row][c] * inv : 0.f;
            sc[row][c] = pv;
            pmax = fmaxf(pmax, pv);
        }
        if (is_last)
            attn_out[((size_t)(b * HEADS_ + h)) * S_ + qrow] = pmax;
    }
    __syncthreads();

    // write p tile
    const size_t pbase = (((size_t)(b * HEADS_ + h)) * S_ + (size_t)qt * 16) * S_;
    for (int idx = tid; idx < 16 * S_; idx += 128) {
        int r = idx >> 9, c = idx & (S_ - 1);
        p[pbase + (size_t)r * S_ + c] = sc[r][c];
    }
}

// ---------------------------------------------------------------------------
// out = LayerNorm(a + b) * g + beta    (one 256-thread block per H=1024 row)
// ---------------------------------------------------------------------------
__global__ __launch_bounds__(256) void add_layernorm_kernel(
    const float* __restrict__ a, const float* __restrict__ bsum,
    const float* __restrict__ g, const float* __restrict__ beta,
    float* __restrict__ out)
{
    int row = blockIdx.x;
    int tid = threadIdx.x;
    const float* pa = a + (size_t)row * H_;
    const float* pb = bsum + (size_t)row * H_;
    __shared__ float red[256];

    float v[4]; float s = 0.f;
    #pragma unroll
    for (int i = 0; i < 4; ++i) { v[i] = pa[tid + 256 * i] + pb[tid + 256 * i]; s += v[i]; }
    red[tid] = s; __syncthreads();
    for (int st = 128; st > 0; st >>= 1) { if (tid < st) red[tid] += red[tid + st]; __syncthreads(); }
    float mean = red[0] * (1.f / (float)H_); __syncthreads();

    float sq = 0.f;
    #pragma unroll
    for (int i = 0; i < 4; ++i) { float d = v[i] - mean; sq += d * d; }
    red[tid] = sq; __syncthreads();
    for (int st = 128; st > 0; st >>= 1) { if (tid < st) red[tid] += red[tid + st]; __syncthreads(); }
    float rstd = rsqrtf(red[0] * (1.f / (float)H_) + 1e-5f);

    #pragma unroll
    for (int i = 0; i < 4; ++i) {
        int c = tid + 256 * i;
        out[(size_t)row * H_ + c] = (v[i] - mean) * rstd * g[c] + beta[c];
    }
}

// ---------------------------------------------------------------------------
// In-place row softmax over N columns (one 256-thread block per row)
// ---------------------------------------------------------------------------
__global__ __launch_bounds__(256) void softmax_rows_kernel(float* __restrict__ logits, int N)
{
    int row = blockIdx.x;
    int tid = threadIdx.x;
    float* pr = logits + (size_t)row * N;
    __shared__ float red[256];

    float mx = -3.4e38f;
    for (int c = tid; c < N; c += 256) mx = fmaxf(mx, pr[c]);
    red[tid] = mx; __syncthreads();
    for (int st = 128; st > 0; st >>= 1) { if (tid < st) red[tid] = fmaxf(red[tid], red[tid + st]); __syncthreads(); }
    mx = red[0]; __syncthreads();

    float sum = 0.f;
    for (int c = tid; c < N; c += 256) { float e = __expf(pr[c] - mx); pr[c] = e; sum += e; }
    red[tid] = sum; __syncthreads();
    for (int st = 128; st > 0; st >>= 1) { if (tid < st) red[tid] += red[tid + st]; __syncthreads(); }
    float inv = 1.f / red[0];
    for (int c = tid; c < N; c += 256) pr[c] *= inv;
}

// ---------------------------------------------------------------------------
extern "C" void kernel_launch(void* const* d_in, const int* in_sizes, int n_in,
                              void* d_out, int out_size, void* d_ws, size_t ws_size,
                              hipStream_t stream)
{
    const int*   x       = (const int*)  d_in[0];
    const int*   lengths = (const int*)  d_in[1];
    const float* emb     = (const float*)d_in[2];
    const float* Wq = (const float*)d_in[3];  const float* bq = (const float*)d_in[4];
    const float* Wk = (const float*)d_in[5];  const float* bk = (const float*)d_in[6];
    const float* Wv = (const float*)d_in[7];  const float* bv = (const float*)d_in[8];
    const float* W1 = (const float*)d_in[9];  const float* b1 = (const float*)d_in[10];
    const float* W2 = (const float*)d_in[11]; const float* b2 = (const float*)d_in[12];
    const float* g1 = (const float*)d_in[13]; const float* be1 = (const float*)d_in[14];
    const float* g2 = (const float*)d_in[15]; const float* be2 = (const float*)d_in[16];
    const float* Wfc = (const float*)d_in[17]; const float* bfc = (const float*)d_in[18];

    float* yout = (float*)d_out;                       // [M_, V_] y_prim
    float* attn_out = yout + (size_t)M_ * V_;          // [B_, HEADS_, S_]

    // workspace carve
    const size_t MH  = (size_t)M_ * H_;                // 8 Mi elements
    const size_t LHH = (size_t)L_ * H_ * H_;           // per weight stack
    float* ws = (float*)d_ws;
    float* z  = ws;            // [M,H]
    float* z1 = z  + MH;
    float* qb = z1 + MH;       // also FFN t1 / out_fc
    float* kb = qb + MH;       // also FFN t2
    float* vb = kb + MH;
    float* ob = vb + MH;       // attention output, flat [B,H,S,dh] == [B,S,H]
    float* pb = ob + MH;       // [B,H,S,S] probs
    // bf16 weight pool (one-time conversion per call)
    bf16* WqB  = (bf16*)(pb + (size_t)B_ * HEADS_ * S_ * S_);
    bf16* WkB  = WqB + LHH;
    bf16* WvB  = WkB + LHH;
    bf16* W1B  = WvB + LHH;
    bf16* W2B  = W1B + LHH;
    bf16* WfcB = W2B + LHH;
    bf16* embB = WfcB + (size_t)H_ * H_;

    // one-time f32 -> bf16 weight conversion
    {
        long long n = (long long)LHH;
        int blocks = (int)((n + 255) / 256);
        cvt_bf16_kernel<<<blocks, 256, 0, stream>>>(Wq, WqB, n);
        cvt_bf16_kernel<<<blocks, 256, 0, stream>>>(Wk, WkB, n);
        cvt_bf16_kernel<<<blocks, 256, 0, stream>>>(Wv, WvB, n);
        cvt_bf16_kernel<<<blocks, 256, 0, stream>>>(W1, W1B, n);
        cvt_bf16_kernel<<<blocks, 256, 0, stream>>>(W2, W2B, n);
        long long nf = (long long)H_ * H_;
        cvt_bf16_kernel<<<(int)((nf + 255) / 256), 256, 0, stream>>>(Wfc, WfcB, nf);
        long long ne = (long long)V_ * H_;
        cvt_bf16_kernel<<<(int)((ne + 255) / 256), 256, 0, stream>>>(emb, embB, ne);
    }

    embed_pe_kernel<<<M_, 256, 0, stream>>>(x, emb, z);

    const dim3 gQKV(H_ / 64, M_ / 128, 1);              // (16,64)
    const dim3 gPV(DH_ / 64, S_ / 64, B_ * HEADS_);     // (4,8,64)
    const dim3 gAttn(S_ / 16, HEADS_, B_);              // (32,4,16)

    for (int l = 0; l < L_; ++l) {
        const size_t wo = (size_t)l * H_ * H_, bo = (size_t)l * H_;
        // q,k,v = z @ W^T + b
        gemm_bf16w_kernel<<<gQKV, 128, 0, stream>>>(
            z, WqB + wo, bq + bo, qb, M_, H_, H_, H_, H_, 0);
        gemm_bf16w_kernel<<<gQKV, 128, 0, stream>>>(
            z, WkB + wo, bk + bo, kb, M_, H_, H_, H_, H_, 0);
        gemm_bf16w_kernel<<<gQKV, 128, 0, stream>>>(
            z, WvB + wo, bv + bo, vb, M_, H_, H_, H_, H_, 0);
        // p = masked softmax(q k^T / sqrt(H))
        attn_scores_kernel<<<gAttn, 128, 0, stream>>>(
            qb, kb, lengths, pb, attn_out, (l == L_ - 1) ? 1 : 0);
        // o = p @ v   (batched over b,h; v head slice has row stride H)
        gemm_pv_kernel<<<gPV, 128, 0, stream>>>(
            pb, vb, ob, S_, DH_, S_, S_, H_, DH_, HEADS_,
            (long long)HEADS_ * S_ * S_, (long long)S_ * S_,
            (long long)S_ * H_,          (long long)DH_,
            (long long)HEADS_ * S_ * DH_,(long long)S_ * DH_);
        // z1 = LN(z + o)
        add_layernorm_kernel<<<M_, 256, 0, stream>>>(z, ob, g1 + bo, be1 + bo, z1);
        // f = gelu(z1 W1^T + b1) W2^T + b2
        gemm_bf16w_kernel<<<gQKV, 128, 0, stream>>>(
            z1, W1B + wo, b1 + bo, qb, M_, H_, H_, H_, H_, 1);
        gemm_bf16w_kernel<<<gQKV, 128, 0, stream>>>(
            qb, W2B + wo, b2 + bo, kb, M_, H_, H_, H_, H_, 0);
        // z = LN(z1 + f)
        add_layernorm_kernel<<<M_, 256, 0, stream>>>(z1, kb, g2 + bo, be2 + bo, z);
    }

    // out_fc = z Wfc^T + bfc
    gemm_bf16w_kernel<<<gQKV, 128, 0, stream>>>(
        z, WfcB, bfc, qb, M_, H_, H_, H_, H_, 0);
    // logits = out_fc @ emb^T  (tied projection) straight into d_out
    gemm_bf16w_kernel<<<dim3(V_ / 64, M_ / 128, 1), 128, 0, stream>>>(
        qb, embB, nullptr, yout, M_, V_, H_, H_, V_, 0);
    softmax_rows_kernel<<<M_, 256, 0, stream>>>(yout, V_);
}
